// FFT_Net_18811956756735
// MI455X (gfx1250) — matrix operand order repaired and checked by reading, not compiled, not run on hardware
//
#include <hip/hip_runtime.h>
#include <stdint.h>

// ---------------------------------------------------------------------------
// MI455X (gfx1250) fused 2-D complex DFT + InstanceNorm.
//   Z = W512 * X * W256  (complex, per (b,c) plane), then per-channel norm.
// bf16x3 split-precision GEMMs on v_wmma_f32_16x16x32_bf16 (compute-bound:
// ~500 FLOP/byte >> fp32 balance; bf16 matrix pipes are the fast path).
// LDS tiles that are pure copies are fed by global_load_async_to_lds_b128
// (ASYNCcnt DMA path) instead of VGPR round-trips.
// ---------------------------------------------------------------------------

typedef __bf16 bf16;
typedef bf16  v16bf __attribute__((ext_vector_type(16)));
typedef float v8f   __attribute__((ext_vector_type(8)));
typedef float f32x4 __attribute__((ext_vector_type(4)));
typedef int   v4i   __attribute__((ext_vector_type(4)));

#define LDK 40          // LDS row stride (elements); 80 B, 16-B aligned rows
#define PLANES 256      // B*C = 16*16
#define RDIM 512
#define DDIM 256
#define PL ((size_t)PLANES * RDIM * DDIM)   // elements per Y plane

#if __has_builtin(__builtin_amdgcn_global_load_async_to_lds_b128)
#define HAVE_ASYNC_LDS 1
#else
#define HAVE_ASYNC_LDS 0
#endif

typedef __attribute__((address_space(1))) v4i gv4i;   // global 16-B chunk
typedef __attribute__((address_space(3))) v4i lv4i;   // LDS 16-B chunk

// Async DMA of 16 bytes global -> LDS (ASYNCcnt). AS pointers are built via
// integer casts: AS1 value == flat value; LDS offset == flat addr[31:0]
// (ISA 10.2 aperture rule), so these casts are exact.
__device__ __forceinline__ void async_copy_b128(const bf16* g, bf16* l) {
#if HAVE_ASYNC_LDS
    __builtin_amdgcn_global_load_async_to_lds_b128(
        (gv4i*)(uintptr_t)g, (lv4i*)(uint32_t)(uintptr_t)l, 0, 0);
#else
    *(f32x4*)l = *(const f32x4*)g;
#endif
}

__device__ __forceinline__ void wait_async() {
#if HAVE_ASYNC_LDS
#if __has_builtin(__builtin_amdgcn_s_wait_asynccnt)
    __builtin_amdgcn_s_wait_asynccnt(0);
#else
    asm volatile("s_wait_asynccnt 0" ::: "memory");
#endif
#endif
}

__device__ __forceinline__ void bfsplit(float x, bf16& hi, bf16& lo) {
    bf16 h = (bf16)x;
    hi = h;
    lo = (bf16)(x - (float)h);
}

__device__ __forceinline__ v8f wmma_bf16(v16bf a, v16bf b, v8f c) {
    // D = A(16x32 bf16) * B(32x16 bf16) + C(16x16 f32)
    return __builtin_amdgcn_wmma_f32_16x16x32_bf16(
        false, a, false, b, (short)0, c, false, false);
}

// A fragment: lane L holds row m=L&15; K = kbase + 0..15, kbase=(L<16)?0:16.
__device__ __forceinline__ v16bf frag_a(const bf16* tile, int m, int kbase) {
    const bf16* p = tile + m * LDK + kbase;
    union { v16bf v; f32x4 q[2]; } u;
    u.q[0] = *(const f32x4*)(p);
    u.q[1] = *(const f32x4*)(p + 8);
    return u.v;
}

// B fragment (tile stored transposed [n][k]): lane L holds col n=L&15;
// K = kbase + 0..15, kbase = (L<16)?0:16.
__device__ __forceinline__ v16bf frag_b(const bf16* tile, int n, int kbase) {
    const bf16* p = tile + n * LDK + kbase;
    union { v16bf v; f32x4 q[2]; } u;
    u.q[0] = *(const f32x4*)(p);
    u.q[1] = *(const f32x4*)(p + 8);
    return u.v;
}

// ---------------------------------------------------------------------------
// Pass 0a: zero the per-(b,ch) {sum, sumsq} accumulators (16*32*2 floats).
// ---------------------------------------------------------------------------
__global__ void fftnet_zero_accum(float* __restrict__ accum) {
    accum[threadIdx.x] = 0.0f;
}

// ---------------------------------------------------------------------------
// Pass 0b: split an f32 matrix into bf16 {hi, lo} planes (done once; the DFT
// weights then stream to LDS via async DMA with zero VALU in the GEMM loops).
// ---------------------------------------------------------------------------
__global__ __launch_bounds__(256) void fftnet_split_w(
    const float* __restrict__ src, bf16* __restrict__ dh, bf16* __restrict__ dl)
{
    size_t i = (size_t)blockIdx.x * 256 + threadIdx.x;
    bf16 h, l;
    bfsplit(src[i], h, l);
    dh[i] = h;
    dl[i] = l;
}

// ---------------------------------------------------------------------------
// Pass 1: Y = W512 * X per plane.  M=512(h) K=512(r) N=256(d).
// Grid (N/32, M/64, PLANES), 256 threads (8 waves, 4x2 of 16x16 tiles).
// A tiles (pre-split W512) arrive via async DMA; B tiles (X) are converted.
// Output: Y split to bf16 hi/lo per component (4 planes in workspace).
// ---------------------------------------------------------------------------
__global__ __launch_bounds__(256) void fftnet_pass1_range_dft(
    const float* __restrict__ xr, const float* __restrict__ xi,
    const bf16* __restrict__ w512,   // 4 planes of 512*512: {Wr_hi,Wr_lo,Wi_hi,Wi_lo}
    bf16* __restrict__ ybase)        // 4 planes of PL:      {Yr_hi,Yr_lo,Yi_hi,Yi_lo}
{
    __shared__ __align__(16) bf16 aT[4][64 * LDK];   // {rH,rL,iH,iL}
    __shared__ __align__(16) bf16 bX[4][32 * LDK];   // {rH,rL,iH,iL} transposed [n][k]

    const int tid  = threadIdx.x;
    const int lane = tid & 31;
    const int wave = tid >> 5;
    const int wm   = wave >> 1;       // 0..3
    const int wn   = wave & 1;        // 0..1
    const int plane = blockIdx.z;
    const int h0 = blockIdx.y * 64;
    const int n0 = blockIdx.x * 32;

    const float* xrP = xr + (size_t)plane * RDIM * DDIM;
    const float* xiP = xi + (size_t)plane * RDIM * DDIM;

    const int am  = wm * 16 + (lane & 15);
    const int akb = (lane < 16) ? 0 : 16;
    const int bn  = wn * 16 + (lane & 15);
    const int bkb = (lane < 16) ? 0 : 16;

    v8f c_yr = {}, c_t = {}, c_yi = {};

    for (int k0 = 0; k0 < RDIM; k0 += 32) {
        __syncthreads();
        // A fill: 4 x (64x32) bf16 tiles of pre-split W512 -> async DMA.
        // 1024 x 16B chunks, 4 per thread.
        #pragma unroll
        for (int i = 0; i < 4; ++i) {
            int id = tid + i * 256;
            int m = id >> 8, r = id & 255;
            int hh = r >> 2, kk8 = (r & 3) * 8;
            async_copy_b128(w512 + ((size_t)m << 18)
                                 + (size_t)(h0 + hh) * RDIM + (k0 + kk8),
                            &aT[m][hh * LDK + kk8]);
        }
        // B fill: X 32(k=r) x 32(n=d) tile, f32 -> bf16 hi/lo, stored [n][k].
        #pragma unroll
        for (int i = 0; i < 4; ++i) {
            int e = tid + i * 256;
            int kk = e >> 5, nn = e & 31;
            float fr = xrP[(size_t)(k0 + kk) * DDIM + (n0 + nn)];
            float fi = xiP[(size_t)(k0 + kk) * DDIM + (n0 + nn)];
            bf16 h, l;
            bfsplit(fr, h, l); bX[0][nn * LDK + kk] = h; bX[1][nn * LDK + kk] = l;
            bfsplit(fi, h, l); bX[2][nn * LDK + kk] = h; bX[3][nn * LDK + kk] = l;
        }
        wait_async();            // this wave's DMA landed in LDS
        __syncthreads();         // publish all waves' tiles

        v16bf arH = frag_a(aT[0], am, akb), arL = frag_a(aT[1], am, akb);
        v16bf aiH = frag_a(aT[2], am, akb), aiL = frag_a(aT[3], am, akb);
        v16bf brH = frag_b(bX[0], bn, bkb), brL = frag_b(bX[1], bn, bkb);
        v16bf biH = frag_b(bX[2], bn, bkb), biL = frag_b(bX[3], bn, bkb);

        // Yr_acc += Wr*Xr ; T_acc += Wi*Xi ; Yi_acc += Wr*Xi + Wi*Xr (bf16x3 each)
        c_yr = wmma_bf16(arH, brH, c_yr);
        c_yr = wmma_bf16(arH, brL, c_yr);
        c_yr = wmma_bf16(arL, brH, c_yr);
        c_t  = wmma_bf16(aiH, biH, c_t);
        c_t  = wmma_bf16(aiH, biL, c_t);
        c_t  = wmma_bf16(aiL, biH, c_t);
        c_yi = wmma_bf16(arH, biH, c_yi);
        c_yi = wmma_bf16(arH, biL, c_yi);
        c_yi = wmma_bf16(arL, biH, c_yi);
        c_yi = wmma_bf16(aiH, brH, c_yi);
        c_yi = wmma_bf16(aiH, brL, c_yi);
        c_yi = wmma_bf16(aiL, brH, c_yi);
    }

    // Epilogue: Yr = c_yr - c_t, Yi = c_yi; store split bf16 (4 planes).
    const int d     = n0 + wn * 16 + (lane & 15);
    const int hbase = h0 + wm * 16 + ((lane >> 4) << 3);
    #pragma unroll
    for (int e = 0; e < 8; ++e) {
        int h = hbase + e;
        size_t idx = (size_t)plane * RDIM * DDIM + (size_t)h * DDIM + d;
        float yr = c_yr[e] - c_t[e];
        float yi = c_yi[e];
        bf16 hh, ll;
        bfsplit(yr, hh, ll); ybase[idx] = hh;          ybase[idx + PL]     = ll;
        bfsplit(yi, hh, ll); ybase[idx + 2 * PL] = hh; ybase[idx + 3 * PL] = ll;
    }
}

// ---------------------------------------------------------------------------
// Pass 2: Z = Y * W256 per plane.  M=512(h) K=256(d) N=256(k_out).
// Both A (split Y) and B (split W256) tiles arrive via async DMA.  B exploits
// DFT-matrix symmetry (W256^T == W256): rows copy straight into the
// transposed-[n][k] LDS layout.  Writes raw Z into d_out + channel stats.
// ---------------------------------------------------------------------------
__global__ __launch_bounds__(256) void fftnet_pass2_doppler_dft(
    const bf16* __restrict__ ybase,  // 4 planes of PL
    const bf16* __restrict__ w256,   // 4 planes of 256*256
    float* __restrict__ out, float* __restrict__ accum)
{
    __shared__ __align__(16) bf16 aT[4][64 * LDK];
    __shared__ __align__(16) bf16 bT[4][32 * LDK];
    __shared__ float red[4];

    const int tid  = threadIdx.x;
    const int lane = tid & 31;
    const int wave = tid >> 5;
    const int wm   = wave >> 1;
    const int wn   = wave & 1;
    const int plane = blockIdx.z;
    const int h0 = blockIdx.y * 64;
    const int n0 = blockIdx.x * 32;

    if (tid < 4) red[tid] = 0.0f;

    const size_t yoff = (size_t)plane * RDIM * DDIM;

    const int am  = wm * 16 + (lane & 15);
    const int akb = (lane < 16) ? 0 : 16;
    const int bn  = wn * 16 + (lane & 15);
    const int bkb = (lane < 16) ? 0 : 16;

    v8f c_zr = {}, c_t = {}, c_zi = {};

    for (int k0 = 0; k0 < DDIM; k0 += 32) {
        __syncthreads();
        // A fill: 4 x (64x32) bf16 tiles of split Y -> async DMA (4/thread).
        #pragma unroll
        for (int i = 0; i < 4; ++i) {
            int id = tid + i * 256;
            int m = id >> 8, r = id & 255;
            int hh = r >> 2, kk8 = (r & 3) * 8;
            async_copy_b128(ybase + (size_t)m * PL + yoff
                                  + (size_t)(h0 + hh) * DDIM + (k0 + kk8),
                            &aT[m][hh * LDK + kk8]);
        }
        // B fill: 4 x (32x32) bf16 tiles of split W256 -> async DMA (2/thread).
        // W256 symmetric: row (n0+nn) supplies column (n0+nn) of B.
        #pragma unroll
        for (int i = 0; i < 2; ++i) {
            int id = tid + i * 256;
            int m = id >> 7, r = id & 127;
            int nn = r >> 2, kk8 = (r & 3) * 8;
            async_copy_b128(w256 + ((size_t)m << 16)
                                 + (size_t)(n0 + nn) * DDIM + (k0 + kk8),
                            &bT[m][nn * LDK + kk8]);
        }
        wait_async();
        __syncthreads();

        v16bf arH = frag_a(aT[0], am, akb), arL = frag_a(aT[1], am, akb);
        v16bf aiH = frag_a(aT[2], am, akb), aiL = frag_a(aT[3], am, akb);
        v16bf brH = frag_b(bT[0], bn, bkb), brL = frag_b(bT[1], bn, bkb);
        v16bf biH = frag_b(bT[2], bn, bkb), biL = frag_b(bT[3], bn, bkb);

        // Zr_acc += Yr*Wr ; T_acc += Yi*Wi ; Zi_acc += Yr*Wi + Yi*Wr
        c_zr = wmma_bf16(arH, brH, c_zr);
        c_zr = wmma_bf16(arH, brL, c_zr);
        c_zr = wmma_bf16(arL, brH, c_zr);
        c_t  = wmma_bf16(aiH, biH, c_t);
        c_t  = wmma_bf16(aiH, biL, c_t);
        c_t  = wmma_bf16(aiL, biH, c_t);
        c_zi = wmma_bf16(arH, biH, c_zi);
        c_zi = wmma_bf16(arH, biL, c_zi);
        c_zi = wmma_bf16(arL, biH, c_zi);
        c_zi = wmma_bf16(aiH, brH, c_zi);
        c_zi = wmma_bf16(aiH, brL, c_zi);
        c_zi = wmma_bf16(aiL, brH, c_zi);
    }

    // Epilogue: raw Z -> d_out (zr -> ch c, zi -> ch 16+c) + channel stats.
    const int b = plane >> 4;
    const int c = plane & 15;
    const int n     = n0 + wn * 16 + (lane & 15);
    const int hbase = h0 + wm * 16 + ((lane >> 4) << 3);

    float sr = 0.f, sr2 = 0.f, si = 0.f, si2 = 0.f;
    #pragma unroll
    for (int e = 0; e < 8; ++e) {
        int h = hbase + e;
        float zr = c_zr[e] - c_t[e];
        float zi = c_zi[e];
        out[((size_t)(b * 32 + c)      * RDIM + h) * DDIM + n] = zr;
        out[((size_t)(b * 32 + 16 + c) * RDIM + h) * DDIM + n] = zi;
        sr += zr; sr2 += zr * zr;
        si += zi; si2 += zi * zi;
    }
    __syncthreads();           // red[] initialized; tiles no longer needed
    atomicAdd(&red[0], sr);
    atomicAdd(&red[1], sr2);
    atomicAdd(&red[2], si);
    atomicAdd(&red[3], si2);
    __syncthreads();
    if (tid == 0) {
        atomicAdd(&accum[(size_t)(b * 32 + c)      * 2 + 0], red[0]);
        atomicAdd(&accum[(size_t)(b * 32 + c)      * 2 + 1], red[1]);
        atomicAdd(&accum[(size_t)(b * 32 + 16 + c) * 2 + 0], red[2]);
        atomicAdd(&accum[(size_t)(b * 32 + 16 + c) * 2 + 1], red[3]);
    }
}

// ---------------------------------------------------------------------------
// Pass 3: in-place InstanceNorm over d_out, float4 vectorized.
// 512 planes of 512*256; 32768 float4 per plane.
// ---------------------------------------------------------------------------
__global__ __launch_bounds__(256) void fftnet_pass3_norm(
    float* __restrict__ out, const float* __restrict__ accum)
{
    const unsigned gid = blockIdx.x * 256u + threadIdx.x;   // 0 .. 2^24-1
    const unsigned plane = gid >> 15;                       // b*32 + ch
    const float invN = 1.0f / (float)(RDIM * DDIM);
    float s  = accum[(size_t)plane * 2 + 0];
    float sq = accum[(size_t)plane * 2 + 1];
    float mean = s * invN;
    float var  = sq * invN - mean * mean;
    float scale = rsqrtf(var + 1e-5f);

    f32x4* p = (f32x4*)out + gid;
    f32x4 v = *p;
    v.x = (v.x - mean) * scale;
    v.y = (v.y - mean) * scale;
    v.z = (v.z - mean) * scale;
    v.w = (v.w - mean) * scale;
    *p = v;
}

// ---------------------------------------------------------------------------
extern "C" void kernel_launch(void* const* d_in, const int* in_sizes, int n_in,
                              void* d_out, int out_size, void* d_ws, size_t ws_size,
                              hipStream_t stream)
{
    const float* xr    = (const float*)d_in[0];
    const float* xi    = (const float*)d_in[1];
    const float* Wr512 = (const float*)d_in[2];
    const float* Wi512 = (const float*)d_in[3];
    const float* Wr256 = (const float*)d_in[4];
    const float* Wi256 = (const float*)d_in[5];
    float* out = (float*)d_out;

    // Workspace layout:
    //   [0, 4096)                     : per-(b,ch) {sum,sumsq} accumulators
    //   [4096, +4*PL*2)               : Y planes {Yr_hi,Yr_lo,Yi_hi,Yi_lo} (bf16)
    //   then 4*512*512 bf16           : split W512 {Wr_hi,Wr_lo,Wi_hi,Wi_lo}
    //   then 4*256*256 bf16           : split W256 {Wr_hi,Wr_lo,Wi_hi,Wi_lo}
    float* accum = (float*)d_ws;
    bf16* ybase = (bf16*)((char*)d_ws + 4096);
    bf16* w512s = ybase + 4 * PL;
    bf16* w256s = w512s + 4 * (size_t)(512 * 512);

    fftnet_zero_accum<<<1, 1024, 0, stream>>>(accum);

    fftnet_split_w<<<1024, 256, 0, stream>>>(Wr512, w512s,                w512s + 262144);
    fftnet_split_w<<<1024, 256, 0, stream>>>(Wi512, w512s + 2 * 262144,   w512s + 3 * 262144);
    fftnet_split_w<<<256,  256, 0, stream>>>(Wr256, w256s,                w256s + 65536);
    fftnet_split_w<<<256,  256, 0, stream>>>(Wi256, w256s + 2 * 65536,    w256s + 3 * 65536);

    dim3 grid(DDIM / 32, RDIM / 64, PLANES);   // (8, 8, 256)
    fftnet_pass1_range_dft<<<grid, 256, 0, stream>>>(xr, xi, w512s, ybase);
    fftnet_pass2_doppler_dft<<<grid, 256, 0, stream>>>(ybase, w256s, out, accum);

    fftnet_pass3_norm<<<(16u * 32u * 32768u) / 256u, 256, 0, stream>>>(out, accum);
}
